// GroupedQueryAttentionWithRoPE_3753801417335
// MI455X (gfx1250) — matrix operand order repaired and checked by reading, not compiled, not run on hardware
//
#include <hip/hip_runtime.h>
#include <hip/hip_bf16.h>

// ---------------------------------------------------------------------------
// GQA + RoPE attention layer for gfx1250 (MI455X), wave32, WMMA f16->f32.
// B=4, S=2048, E=2048, HQ=32, HKV=8, HD=64, G=4, scale = 1/sqrt(2048).
// ---------------------------------------------------------------------------

#define Bc   4
#define Sc   2048
#define Ec   2048
#define HQc  32
#define HKVc 8
#define HDc  64
#define Gc   4
#define NKV  (HKVc * HDc)   // 512
#define SCALEc 0.02209708691207961f   // 1/sqrt(2048)
#define LOG2E  1.4426950408889634f

typedef __attribute__((ext_vector_type(16))) _Float16 v16h;
typedef __attribute__((ext_vector_type(8)))  _Float16 v8h;
typedef __attribute__((ext_vector_type(8)))  float    v8f;
typedef __attribute__((vector_size(4 * sizeof(int)))) int v4i;

union V16U { v16h v; v8h h[2]; };

static __device__ __forceinline__ v16h ld2(const _Float16* p0, const _Float16* p1) {
  V16U u;
  u.h[0] = *(const v8h*)p0;
  u.h[1] = *(const v8h*)p1;
  return u.v;
}

static __device__ __forceinline__ v8f wmma16(v16h a, v16h b, v8f c) {
  return __builtin_amdgcn_wmma_f32_16x16x32_f16(false, a, false, b, (short)0, c,
                                                false, false);
}

// ---------------------------------------------------------------------------
// Async global->LDS staging (ASYNCcnt path) with compile-safe fallback.
// Probe-learned signature: arg0 = addrspace(1) v4i* (global src),
// arg1 = addrspace(3) v4i* (LDS dst), then imm offset, imm cpol.
// ---------------------------------------------------------------------------
#if defined(__gfx1250__) && __has_builtin(__builtin_amdgcn_global_load_async_to_lds_b128)
#define ASYNC_LDS 1
#endif

static __device__ __forceinline__ void cp16(_Float16* l, const _Float16* g) {
#ifdef ASYNC_LDS
  __builtin_amdgcn_global_load_async_to_lds_b128(
      (__attribute__((address_space(1))) v4i*)g,
      (__attribute__((address_space(3))) v4i*)l, 0, 0);
#else
  *(v8h*)l = *(const v8h*)g;
#endif
}

static __device__ __forceinline__ void stage_barrier() {
#ifdef ASYNC_LDS
#if __has_builtin(__builtin_amdgcn_s_wait_asynccnt)
  __builtin_amdgcn_s_wait_asynccnt(0);
#else
  asm volatile("s_wait_asynccnt 0x0" ::: "memory");
#endif
#endif
  __syncthreads();
}

// ---------------------------------------------------------------------------
// f32 -> f16 conversion
// ---------------------------------------------------------------------------
__global__ void cvt_f32_f16(const float* __restrict__ in, _Float16* __restrict__ out,
                            long n) {
  long i = (long)blockIdx.x * blockDim.x + threadIdx.x;
  if (i < n) out[i] = (_Float16)in[i];
}

// ---------------------------------------------------------------------------
// GEMM: C[M,N] = A[M,K] (f16) @ W[N,K]^T (f16) + bias[N], fp32 accumulate.
// Block = 4 waves, 128 rows x 64 cols; each wave owns 32 rows x 64 cols
// (8 accumulators, 8 WMMA per 32-deep k-chunk). The 64x32 weight chunk is
// staged in LDS once per chunk (async if available) and shared by all waves.
// ---------------------------------------------------------------------------
template <bool F16OUT>
__global__ void gemm_bias(const _Float16* __restrict__ A,
                          const _Float16* __restrict__ W,
                          const float* __restrict__ bias,
                          void* __restrict__ Cout,
                          int N, int K) {
  __shared__ alignas(16) _Float16 Wt[64 * 32];   // [col_local][k_local]

  const int lane = threadIdx.x & 31;
  const int wv   = threadIdx.x >> 5;
  const int hh   = lane >> 4;        // half-wave (0/1)
  const int n16  = lane & 15;
  const int row0 = blockIdx.y * 128 + wv * 32;   // wave's 32-row strip
  const int cb   = blockIdx.x * 64;

  v8f acc[2][4] = {};
  const _Float16* Arow0 = A + (long)(row0 + n16) * K;       // row tile 0
  const _Float16* Arow1 = A + (long)(row0 + 16 + n16) * K;  // row tile 1

  for (int k0 = 0; k0 < K; k0 += 32) {
    __syncthreads();
    {
      // cooperative stage of 64x32 weight chunk, W[n][k] -> Wt[col][k]
      int cl = threadIdx.x >> 1;          // 0..63
      int kk = (threadIdx.x & 1) * 16;    // 0 or 16
      const _Float16* src = W + (long)(cb + cl) * K + k0 + kk;
      cp16(&Wt[cl * 32 + kk],     src);
      cp16(&Wt[cl * 32 + kk + 8], src + 8);
      if (k0 + 32 < K) __builtin_prefetch(src + 32, 0, 1);  // global_prefetch_b8
    }
    stage_barrier();

    // A fragments (16x32 f16): half-wave hh holds contiguous K [16h..16h+15]
    v16h a0 = ld2(Arow0 + k0 + 16 * hh, Arow0 + k0 + 16 * hh + 8);
    v16h a1 = ld2(Arow1 + k0 + 16 * hh, Arow1 + k0 + 16 * hh + 8);
#pragma unroll
    for (int t = 0; t < 4; ++t) {
      // B fragment (32x16): lane holds col n16, halves = contiguous K at 16*hh
      const _Float16* wp = &Wt[(16 * t + n16) * 32 + 16 * hh];
      v16h b = ld2(wp, wp + 8);
      acc[0][t] = wmma16(a0, b, acc[0][t]);
      acc[1][t] = wmma16(a1, b, acc[1][t]);
    }
  }

  // Epilogue: bias add + store. C layout: VGPR e -> row e + 8*hh, col per lane.
#pragma unroll
  for (int rt = 0; rt < 2; ++rt) {
#pragma unroll
    for (int t = 0; t < 4; ++t) {
      int col = cb + 16 * t + n16;
      float bz = bias[col];
#pragma unroll
      for (int e = 0; e < 8; ++e) {
        int row = row0 + rt * 16 + e + 8 * hh;
        float v = acc[rt][t][e] + bz;
        if constexpr (F16OUT)
          ((_Float16*)Cout)[(long)row * N + col] = (_Float16)v;
        else
          ((float*)Cout)[(long)row * N + col] = v;
      }
    }
  }
}

// ---------------------------------------------------------------------------
// RoPE, in place on a [B][S][H][HD] f16 tensor. One thread per (pair).
// ---------------------------------------------------------------------------
__global__ void rope_f16(_Float16* __restrict__ t,
                         const float* __restrict__ cosT,
                         const float* __restrict__ sinT,
                         int H, long total) {
  long i = (long)blockIdx.x * blockDim.x + threadIdx.x;
  if (i >= total) return;
  const int pairs = HDc / 2;                 // 32
  int pi = (int)(i % pairs);
  long bsh = i / pairs;                      // (b,s,h) flat, h fastest
  int s = (int)((bsh / H) % Sc);
  long base = bsh * HDc + 2 * pi;
  float tr = (float)t[base];
  float ti = (float)t[base + 1];
  float c = cosT[s * pairs + pi];
  float sn = sinT[s * pairs + pi];
  t[base]     = (_Float16)(tr * c - ti * sn);
  t[base + 1] = (_Float16)(tr * sn + ti * c);
}

// ---------------------------------------------------------------------------
// Flash attention, causal, GQA. Block = 4 waves = the 4 Q-heads of one KV
// group; each wave owns a 16-row Q tile. K/V 32-key chunks staged in LDS
// (K async if available, V transposed manually) and shared across the 4
// waves. Online softmax in fp32; P tile transposed via per-wave LDS.
// ---------------------------------------------------------------------------
__global__ void flash_attn(const _Float16* __restrict__ Q,
                           const _Float16* __restrict__ Kb,
                           const _Float16* __restrict__ Vb,
                           _Float16* __restrict__ O) {
  __shared__ alignas(16) _Float16 Kt[32 * 64];      // [key][d]
  __shared__ alignas(16) _Float16 Vt[64 * 32];      // [d][key]  (transposed)
  __shared__ alignas(16) _Float16 Pt[4][16 * 32];   // per-wave P tile

  const int lane = threadIdx.x & 31;
  const int wv   = threadIdx.x >> 5;
  const int hh   = lane >> 4;
  const int n16  = lane & 15;
  const int qbase = blockIdx.x * 16;
  const int kvh   = blockIdx.y;
  const int b     = blockIdx.z;
  const int hq    = kvh * Gc + wv;

  // Q A-fragments for the two d-chunks (d 0..31, 32..63); row = qbase + n16.
  const _Float16* qrow =
      Q + (((long)(b * Sc + qbase + n16)) * HQc + hq) * HDc;
  v16h aq0 = ld2(qrow + 16 * hh,      qrow + 16 * hh + 8);
  v16h aq1 = ld2(qrow + 32 + 16 * hh, qrow + 32 + 16 * hh + 8);

  v8f o[4] = {};
  float mrun[8], lrun[8];
#pragma unroll
  for (int e = 0; e < 8; ++e) { mrun[e] = -3.0e38f; lrun[e] = 0.0f; }

  const int khi = qbase + 15;
  for (int kb = 0; kb <= khi; kb += 32) {
    __syncthreads();
    {
      // cooperative K/V chunk stage: 32 keys x 64 dims each
      int key = threadIdx.x >> 2;           // 0..31
      int c0  = (threadIdx.x & 3) * 16;     // 0,16,32,48
      long gbase = (((long)(b * Sc + kb + key)) * HKVc + kvh) * HDc + c0;
      cp16(&Kt[key * 64 + c0],     Kb + gbase);
      cp16(&Kt[key * 64 + c0 + 8], Kb + gbase + 8);
      v8h v0 = *(const v8h*)(Vb + gbase);
      v8h v1 = *(const v8h*)(Vb + gbase + 8);
#pragma unroll
      for (int j = 0; j < 8; ++j) {
        Vt[(c0 + j) * 32 + key]     = v0[j];
        Vt[(c0 + 8 + j) * 32 + key] = v1[j];
      }
    }
    stage_barrier();

    // S = Q * K^T  (two 16x16 f32 tiles, keys kb..kb+15 and kb+16..kb+31)
    v8f s0 = {}, s1 = {};
    {
      const _Float16* kp = &Kt[n16 * 64];
      s0 = wmma16(aq0, ld2(kp + 16 * hh,      kp + 16 * hh + 8),      s0);
      s0 = wmma16(aq1, ld2(kp + 32 + 16 * hh, kp + 32 + 16 * hh + 8), s0);
      kp = &Kt[(16 + n16) * 64];
      s1 = wmma16(aq0, ld2(kp + 16 * hh,      kp + 16 * hh + 8),      s1);
      s1 = wmma16(aq1, ld2(kp + 32 + 16 * hh, kp + 32 + 16 * hh + 8), s1);
    }

    // scale + causal mask (C layout: element e -> row qbase+e+8*hh, col n16)
#pragma unroll
    for (int e = 0; e < 8; ++e) {
      int qg = qbase + e + 8 * hh;
      s0[e] = (kb + n16      <= qg) ? s0[e] * SCALEc : -3.0e38f;
      s1[e] = (kb + 16 + n16 <= qg) ? s1[e] * SCALEc : -3.0e38f;
    }

    // row max across the 16 lanes of the half-wave
    float mc[8];
#pragma unroll
    for (int e = 0; e < 8; ++e) mc[e] = fmaxf(s0[e], s1[e]);
#pragma unroll
    for (int off = 1; off < 16; off <<= 1) {
#pragma unroll
      for (int e = 0; e < 8; ++e)
        mc[e] = fmaxf(mc[e], __shfl_xor(mc[e], off, 16));
    }

    float alpha[8];
#pragma unroll
    for (int e = 0; e < 8; ++e) {
      float nm = fmaxf(mrun[e], mc[e]);
      alpha[e] = exp2f((mrun[e] - nm) * LOG2E);
      mrun[e] = nm;
      s0[e] = exp2f((s0[e] - nm) * LOG2E);
      s1[e] = exp2f((s1[e] - nm) * LOG2E);
    }

    // row sum
    float rs[8];
#pragma unroll
    for (int e = 0; e < 8; ++e) rs[e] = s0[e] + s1[e];
#pragma unroll
    for (int off = 1; off < 16; off <<= 1) {
#pragma unroll
      for (int e = 0; e < 8; ++e) rs[e] += __shfl_xor(rs[e], off, 16);
    }
#pragma unroll
    for (int e = 0; e < 8; ++e) lrun[e] = lrun[e] * alpha[e] + rs[e];

    // P tile: C layout -> A layout via per-wave LDS round-trip
    _Float16* P = &Pt[wv][0];
#pragma unroll
    for (int e = 0; e < 8; ++e) {
      int r = e + 8 * hh;
      P[r * 32 + n16]      = (_Float16)s0[e];
      P[r * 32 + 16 + n16] = (_Float16)s1[e];
    }

    // rescale running O
#pragma unroll
    for (int t = 0; t < 4; ++t)
#pragma unroll
      for (int e = 0; e < 8; ++e) o[t][e] *= alpha[e];

    // O += P * V
    v16h ap = ld2(&P[n16 * 32 + 16 * hh], &P[n16 * 32 + 16 * hh + 8]);
#pragma unroll
    for (int t = 0; t < 4; ++t) {
      const _Float16* vp = &Vt[(16 * t + n16) * 32 + 16 * hh];
      o[t] = wmma16(ap, ld2(vp, vp + 8), o[t]);
    }
  }

  // normalize + store O as f16 [b][s][hq][d]
#pragma unroll
  for (int e = 0; e < 8; ++e) {
    float inv = 1.0f / lrun[e];
    int qg = qbase + e + 8 * hh;
    long base = (((long)(b * Sc + qg)) * HQc + hq) * HDc;
#pragma unroll
    for (int t = 0; t < 4; ++t)
      O[base + 16 * t + n16] = (_Float16)(o[t][e] * inv);
  }
}

// ---------------------------------------------------------------------------
// Host launch
// ---------------------------------------------------------------------------
extern "C" void kernel_launch(void* const* d_in, const int* in_sizes, int n_in,
                              void* d_out, int out_size, void* d_ws, size_t ws_size,
                              hipStream_t stream) {
  (void)in_sizes; (void)n_in; (void)out_size; (void)ws_size;

  const float* x   = (const float*)d_in[0];
  const float* fc  = (const float*)d_in[1];
  const float* fs  = (const float*)d_in[2];
  const float* wq  = (const float*)d_in[3];
  const float* wqb = (const float*)d_in[4];
  const float* wk  = (const float*)d_in[5];
  const float* wkb = (const float*)d_in[6];
  const float* wv  = (const float*)d_in[7];
  const float* wvb = (const float*)d_in[8];
  const float* wo  = (const float*)d_in[9];
  const float* wob = (const float*)d_in[10];
  float* out = (float*)d_out;

  // workspace layout (100 MB): x16/attnO share the first 32 MB region.
  char* w = (char*)d_ws;
  _Float16* x16  = (_Float16*)(w);
  _Float16* q16  = (_Float16*)(w + (size_t)32  * 1024 * 1024);
  _Float16* k16  = (_Float16*)(w + (size_t)64  * 1024 * 1024);
  _Float16* v16  = (_Float16*)(w + (size_t)72  * 1024 * 1024);
  _Float16* wq16 = (_Float16*)(w + (size_t)80  * 1024 * 1024);
  _Float16* wk16 = (_Float16*)(w + (size_t)88  * 1024 * 1024);
  _Float16* wv16 = (_Float16*)(w + (size_t)90  * 1024 * 1024);
  _Float16* wo16 = (_Float16*)(w + (size_t)92  * 1024 * 1024);
  _Float16* ao16 = x16;  // x16 is dead after the QKV projections

  const long nX  = (long)Bc * Sc * Ec;        // 16.8M
  const long nWq = (long)Ec * Ec;             // 4.2M
  const long nWk = (long)NKV * Ec;            // 1.05M

  cvt_f32_f16<<<dim3((nX  + 255) / 256), 256, 0, stream>>>(x,  x16,  nX);
  cvt_f32_f16<<<dim3((nWq + 255) / 256), 256, 0, stream>>>(wq, wq16, nWq);
  cvt_f32_f16<<<dim3((nWk + 255) / 256), 256, 0, stream>>>(wk, wk16, nWk);
  cvt_f32_f16<<<dim3((nWk + 255) / 256), 256, 0, stream>>>(wv, wv16, nWk);
  cvt_f32_f16<<<dim3((nWq + 255) / 256), 256, 0, stream>>>(wo, wo16, nWq);

  const int M = Bc * Sc;  // 8192
  gemm_bias<true><<<dim3(Ec / 64,  M / 128), 128, 0, stream>>>(x16, wq16, wqb, q16, Ec,  Ec);
  gemm_bias<true><<<dim3(NKV / 64, M / 128), 128, 0, stream>>>(x16, wk16, wkb, k16, NKV, Ec);
  gemm_bias<true><<<dim3(NKV / 64, M / 128), 128, 0, stream>>>(x16, wv16, wvb, v16, NKV, Ec);

  const long nQp = (long)Bc * Sc * HQc * (HDc / 2);
  const long nKp = (long)Bc * Sc * HKVc * (HDc / 2);
  rope_f16<<<dim3((nQp + 255) / 256), 256, 0, stream>>>(q16, fc, fs, HQc,  nQp);
  rope_f16<<<dim3((nKp + 255) / 256), 256, 0, stream>>>(k16, fc, fs, HKVc, nKp);

  flash_attn<<<dim3(Sc / 16, HKVc, Bc), 128, 0, stream>>>(q16, k16, v16, ao16);

  gemm_bias<false><<<dim3(Ec / 64, M / 128), 128, 0, stream>>>(ao16, wo16, wob, out, Ec, Ec);
}
